// WaveletMultiScaleAttention_71657234366663
// MI455X (gfx1250) — compile-verified, matrix operand order checked by reading, not run on hardware
//
#include <hip/hip_runtime.h>

// ---------------------------------------------------------------------------
// WaveletMultiScaleAttention for MI455X (gfx1250, wave32, WMMA bf16)
// B=2, N=2048, D=768, H=12, HD=64, SCALE=1/8
//   1) gemm_bf16<0>: qkv = x @ qkv_w^T + qkv_b   -> split bf16 q/k/v [B,H,N,64]
//   2) attn_kernel : flash attention with multiplicative K_psd on logits
//                    (softmax runs in the exp2 domain: log2e folded into scale)
//   3) gemm_bf16<1>: out = attn @ proj_w^T + proj_b (f32 out)
// Matmuls on v_wmma_f32_16x16x32_bf16; bf16 copies staged with
// global_load_async_to_lds_b128 (ASYNCcnt) where the builtin exists.
// ---------------------------------------------------------------------------

typedef __attribute__((ext_vector_type(16))) __bf16 v16bf;
typedef __attribute__((ext_vector_type(8)))  float  v8f;
typedef unsigned short u16;
typedef unsigned int   u32;
typedef int v4i_vs __attribute__((vector_size(16)));   // matches builtin pointee

#if __has_builtin(__builtin_amdgcn_global_load_async_to_lds_b128)
#define HAVE_ASYNC_LDS 1
#else
#define HAVE_ASYNC_LDS 0
#endif

__device__ __forceinline__ void async_copy16(const void* g, void* lds, int byteOff0) {
#if HAVE_ASYNC_LDS
    // AS1 (global) src / AS3 (LDS) dst; int->AS3 pointer keeps the low 32 bits
    // of the flat address, which is the LDS byte offset the instruction wants.
    __attribute__((address_space(1))) v4i_vs* gp =
        (__attribute__((address_space(1))) v4i_vs*)(size_t)g;
    __attribute__((address_space(3))) v4i_vs* lp =
        (__attribute__((address_space(3))) v4i_vs*)(size_t)lds;
    if (byteOff0 == 0)
        __builtin_amdgcn_global_load_async_to_lds_b128(gp, lp, 0, 0);
    else
        __builtin_amdgcn_global_load_async_to_lds_b128(gp, lp, 16, 0);
#endif
}
__device__ __forceinline__ void async_wait0() {
#if HAVE_ASYNC_LDS
#if __has_builtin(__builtin_amdgcn_s_wait_asynccnt)
    __builtin_amdgcn_s_wait_asynccnt(0);
#else
    asm volatile("s_wait_asynccnt 0x0" ::: "memory");
#endif
#endif
}

__device__ __forceinline__ float fast_exp2(float x) {
#if __has_builtin(__builtin_amdgcn_exp2f)
    return __builtin_amdgcn_exp2f(x);   // single v_exp_f32
#else
    return exp2f(x);
#endif
}

__device__ __forceinline__ u16 f32_to_bf16(float f) {
    u32 x = __builtin_bit_cast(u32, f);
    u32 r = 0x7FFFu + ((x >> 16) & 1u);   // round-to-nearest-even
    return (u16)((x + r) >> 16);
}
__device__ __forceinline__ u32 pack2(float lo, float hi) {
#if __has_builtin(__builtin_amdgcn_cvt_pk_bf16_f32)
    auto r = __builtin_amdgcn_cvt_pk_bf16_f32(lo, hi);   // v_cvt_pk_bf16_f32
    return __builtin_bit_cast(u32, r);
#else
    return (u32)f32_to_bf16(lo) | ((u32)f32_to_bf16(hi) << 16);
#endif
}

union Frag {
    v16bf v;
    uint4 q[2];
    u16   u[16];
};

__device__ __forceinline__ v8f wmma_bf16(const Frag& a, const Frag& b, v8f c) {
    return __builtin_amdgcn_wmma_f32_16x16x32_bf16(
        /*neg_a=*/false, a.v, /*neg_b=*/false, b.v,
        /*c_mod=*/(short)0, c, /*reuse_a=*/false, /*reuse_b=*/false);
}

// load a 16x32 bf16 operand fragment from LDS laid out [row][k] (k contiguous,
// row stride LD ushorts, LD % 8 == 0). lane l: row = base_row + (l&15),
// elements 0..7 <- k = 8*(l>>4)+0..7, elements 8..15 <- k = 16+8*(l>>4)+0..7
__device__ __forceinline__ void load_frag(Frag& f, const u16* p) {
    f.q[0] = *(const uint4*)(p);
    f.q[1] = *(const uint4*)(p + 16);
}

// ======================= GEMM: C = A @ W^T + bias ==========================
// MODE 0: A = f32 x[4096,768], NC=2304, epilogue splits into bf16 q/k/v
// MODE 1: A = bf16 attn[4096,768], NC=768, epilogue writes f32 d_out
template<int MODE>
__global__ __launch_bounds__(256) void gemm_bf16(
    const void* __restrict__ Aptr, const float* __restrict__ W,
    const float* __restrict__ bias,
    u16* __restrict__ outQ, u16* __restrict__ outK, u16* __restrict__ outV,
    float* __restrict__ outF)
{
    constexpr int K = 768, KC = 32, LDA = 40;   // 32 + 8 pad (16B-aligned rows)
    __shared__ u16 As[128 * LDA];
    __shared__ u16 Bs[128 * LDA];

    const int tid  = threadIdx.x;
    const int lane = tid & 31, wave = tid >> 5;
    const int hi   = lane >> 4, l15 = lane & 15;
    const int waveM = wave & 1, waveN = wave >> 1;         // 2 x 4 wave grid
    const int rowBlk = blockIdx.y * 128, colBlk = blockIdx.x * 128;

    v8f acc[4][2];
    #pragma unroll
    for (int mt = 0; mt < 4; ++mt)
        #pragma unroll
        for (int nt = 0; nt < 2; ++nt) acc[mt][nt] = {};

    // staging: each thread owns 16 contiguous K elements of one row
    const int sr = tid >> 1;            // 0..127
    const int sc = (tid & 1) * 16;      // 0 or 16
    const float* wRow = W + (size_t)(colBlk + sr) * K + sc;
    u16* aDst = &As[sr * LDA + sc];
    u16* bDst = &Bs[sr * LDA + sc];

    const float* aRowF = nullptr;
    const u16*   aRowH = nullptr;
    if constexpr (MODE == 0) aRowF = (const float*)Aptr + (size_t)(rowBlk + sr) * K + sc;
    else                     aRowH = (const u16*)Aptr   + (size_t)(rowBlk + sr) * K + sc;

    for (int kk = 0; kk < K; kk += KC) {
        uint4 ap0, ap1, bp0, bp1;
        if constexpr (MODE == 0) {
            const float4* a4 = (const float4*)(aRowF + kk);
            float4 a0 = a4[0], a1 = a4[1], a2 = a4[2], a3 = a4[3];
            ap0 = uint4{pack2(a0.x,a0.y), pack2(a0.z,a0.w), pack2(a1.x,a1.y), pack2(a1.z,a1.w)};
            ap1 = uint4{pack2(a2.x,a2.y), pack2(a2.z,a2.w), pack2(a3.x,a3.y), pack2(a3.z,a3.w)};
        } else if constexpr (!HAVE_ASYNC_LDS) {
            const uint4* a4 = (const uint4*)(aRowH + kk);
            ap0 = a4[0]; ap1 = a4[1];
        }
        {
            const float4* b4 = (const float4*)(wRow + kk);
            float4 b0 = b4[0], b1 = b4[1], b2 = b4[2], b3 = b4[3];
            bp0 = uint4{pack2(b0.x,b0.y), pack2(b0.z,b0.w), pack2(b1.x,b1.y), pack2(b1.z,b1.w)};
            bp1 = uint4{pack2(b2.x,b2.y), pack2(b2.z,b2.w), pack2(b3.x,b3.y), pack2(b3.z,b3.w)};
        }
        __syncthreads();                      // WAR on previous chunk reads
        if constexpr (MODE == 1 && HAVE_ASYNC_LDS) {
            // bf16 A tile: DMA straight into LDS (ASYNCcnt-tracked)
            async_copy16(aRowH + kk, aDst, 0);
            async_copy16(aRowH + kk, aDst, 16);
        } else {
            *(uint4*)(aDst)     = ap0;
            *(uint4*)(aDst + 8) = ap1;
        }
        *(uint4*)(bDst)     = bp0;
        *(uint4*)(bDst + 8) = bp1;
        if constexpr (MODE == 1) async_wait0();
        __syncthreads();                      // RAW: tiles visible

        Frag af[4], bf[2];
        #pragma unroll
        for (int mt = 0; mt < 4; ++mt)
            load_frag(af[mt], &As[(waveM*64 + mt*16 + l15) * LDA + 8*hi]);
        #pragma unroll
        for (int nt = 0; nt < 2; ++nt)
            load_frag(bf[nt], &Bs[(waveN*32 + nt*16 + l15) * LDA + 8*hi]);
        #pragma unroll
        for (int mt = 0; mt < 4; ++mt)
            #pragma unroll
            for (int nt = 0; nt < 2; ++nt)
                acc[mt][nt] = wmma_bf16(af[mt], bf[nt], acc[mt][nt]);
    }

    // epilogue: C element (vgpr j, lane) -> m = j + 8*hi, n = l15
    #pragma unroll
    for (int mt = 0; mt < 4; ++mt) {
        #pragma unroll
        for (int nt = 0; nt < 2; ++nt) {
            const int c  = colBlk + waveN*32 + nt*16 + l15;
            const int r0 = rowBlk + waveM*64 + mt*16 + 8*hi;
            const float bb = bias[c];
            #pragma unroll
            for (int j = 0; j < 8; ++j) {
                const float v = acc[mt][nt][j] + bb;
                const int r = r0 + j;
                if constexpr (MODE == 0) {
                    const int which = c / 768, rem = c - which * 768;
                    const int h = rem >> 6, hd = rem & 63;
                    const int b = r >> 11, n = r & 2047;
                    u16* dst = (which == 0) ? outQ : (which == 1 ? outK : outV);
                    dst[(((size_t)(b*12 + h)) * 2048 + n) * 64 + hd] = f32_to_bf16(v);
                } else {
                    outF[(size_t)r * 768 + c] = v;
                }
            }
        }
    }
}

// ======================= Flash attention with PSD kernel ====================
// grid: (N/128, B*H). 8 waves/block; wave w owns 16 query rows.
// Softmax in exp2 domain: logits pre-scaled by SCALE*log2(e) (exact: the PSD
// kernel is multiplicative), so alpha/P are single v_exp_f32 ops.
__global__ __launch_bounds__(256) void attn_kernel(
    const u16* __restrict__ qb, const u16* __restrict__ kb,
    const u16* __restrict__ vb, const float* __restrict__ Kpsd,
    u16* __restrict__ outA)
{
    constexpr int HD = 64, NSEQ = 2048, LDK = 72;   // 64 + 8 pad
    constexpr float SCALE_LOG2E = 0.125f * 1.44269504088896340736f;
    __shared__ u16 Ks[64 * LDK];          // [key][hd]
    __shared__ u16 Vt[64 * LDK];          // [hd][key]  (transposed V chunk)
    __shared__ u16 Ps[8 * 16 * LDK];      // per-wave P tile [16][LDK]

    const int tid  = threadIdx.x;
    const int lane = tid & 31, wave = tid >> 5;
    const int hi   = lane >> 4, l15 = lane & 15;
    const int bh    = blockIdx.y;                  // 0..23  (= b*12 + h)
    const int qrow0 = blockIdx.x * 128 + wave * 16;

    const u16* qp = qb + (size_t)bh * NSEQ * HD;
    const u16* kp = kb + (size_t)bh * NSEQ * HD;
    const u16* vp = vb + (size_t)bh * NSEQ * HD;

    // Q fragments (A operand), loaded once straight from global
    Frag qf[2];
    #pragma unroll
    for (int kt = 0; kt < 2; ++kt)
        load_frag(qf[kt], qp + (size_t)(qrow0 + l15) * HD + kt*32 + 8*hi);

    float row_max[8], row_sum[8];
    v8f oacc[4];
    #pragma unroll
    for (int j = 0; j < 8; ++j) { row_max[j] = -3.0e38f; row_sum[j] = 0.f; }
    #pragma unroll
    for (int t = 0; t < 4; ++t) oacc[t] = {};

    u16* Pw = &Ps[wave * 16 * LDK];

    // staging addressing: thread t -> key row tid>>2, 16 hd elems at (tid&3)*16
    const int srow = tid >> 2;
    const int scol = (tid & 3) * 16;

    for (int kc = 0; kc < NSEQ / 64; ++kc) {
        __syncthreads();                  // previous chunk fully consumed
        {
            const u16* ksrc = kp + (size_t)(kc*64 + srow) * HD + scol;
            u16* kdst = &Ks[srow * LDK + scol];
            if constexpr (HAVE_ASYNC_LDS) {
                async_copy16(ksrc, kdst, 0);      // ASYNCcnt path
                async_copy16(ksrc, kdst, 16);
            } else {
                const uint4* k4 = (const uint4*)ksrc;
                *(uint4*)(kdst)     = k4[0];
                *(uint4*)(kdst + 8) = k4[1];
            }

            union { uint4 q[2]; u16 u[16]; } vv;
            const uint4* v4 = (const uint4*)(vp + (size_t)(kc*64 + srow) * HD + scol);
            vv.q[0] = v4[0]; vv.q[1] = v4[1];
            #pragma unroll
            for (int e = 0; e < 16; ++e)
                Vt[(scol + e) * LDK + srow] = vv.u[e];   // transpose into LDS
        }
        if (kc + 1 < NSEQ / 64) {          // warm L2/L0 for next chunk
            __builtin_prefetch(kp + (size_t)(kc + 1) * 64 * HD + tid * 16, 0, 1);
            __builtin_prefetch(vp + (size_t)(kc + 1) * 64 * HD + tid * 16, 0, 1);
        }
        async_wait0();
        __syncthreads();

        // S2 = (Q K^T) * SCALE*log2e * K_psd   (16 x 64 per wave, exp2 domain)
        float sreg[4][8];
        #pragma unroll
        for (int nt = 0; nt < 4; ++nt) {
            v8f s = {};
            #pragma unroll
            for (int kt = 0; kt < 2; ++kt) {
                Frag kf;
                load_frag(kf, &Ks[(nt*16 + l15) * LDK + kt*32 + 8*hi]);
                s = wmma_bf16(qf[kt], kf, s);
            }
            const int kcol = kc*64 + nt*16 + l15;
            const int qrg  = qrow0 + 8*hi;
            #pragma unroll
            for (int j = 0; j < 8; ++j) {
                const float ps = Kpsd[(size_t)(qrg + j) * NSEQ + kcol];
                sreg[nt][j] = s[j] * SCALE_LOG2E * ps;
            }
        }

        // online softmax update (rows replicated across each 16-lane half)
        float alpha[8], csum[8];
        #pragma unroll
        for (int j = 0; j < 8; ++j) {
            float m = fmaxf(fmaxf(sreg[0][j], sreg[1][j]),
                            fmaxf(sreg[2][j], sreg[3][j]));
            #pragma unroll
            for (int msk = 1; msk < 16; msk <<= 1)
                m = fmaxf(m, __shfl_xor(m, msk, 32));
            const float nm = fmaxf(row_max[j], m);
            alpha[j]   = fast_exp2(row_max[j] - nm);
            row_max[j] = nm;
            csum[j]    = 0.f;
        }
        #pragma unroll
        for (int nt = 0; nt < 4; ++nt) {
            #pragma unroll
            for (int j = 0; j < 8; ++j) {
                const float p = fast_exp2(sreg[nt][j] - row_max[j]);
                csum[j] += p;
                Pw[(j + 8*hi) * LDK + nt*16 + l15] = f32_to_bf16(p);
            }
        }
        #pragma unroll
        for (int j = 0; j < 8; ++j) {
            float su = csum[j];
            #pragma unroll
            for (int msk = 1; msk < 16; msk <<= 1)
                su += __shfl_xor(su, msk, 32);
            row_sum[j] = row_sum[j] * alpha[j] + su;
        }
        #pragma unroll
        for (int t = 0; t < 4; ++t)
            #pragma unroll
            for (int j = 0; j < 8; ++j)
                oacc[t][j] *= alpha[j];

        // intra-wave: P stores must land before cross-lane A-fragment reload
        asm volatile("s_wait_dscnt 0" ::: "memory");

        // O += P @ V   (contraction over 64 keys, 4 hd tiles)
        Frag pa[2];
        #pragma unroll
        for (int kt = 0; kt < 2; ++kt)
            load_frag(pa[kt], &Pw[l15 * LDK + kt*32 + 8*hi]);
        #pragma unroll
        for (int t = 0; t < 4; ++t) {
            #pragma unroll
            for (int kt = 0; kt < 2; ++kt) {
                Frag vf;
                load_frag(vf, &Vt[(t*16 + l15) * LDK + kt*32 + 8*hi]);
                oacc[t] = wmma_bf16(pa[kt], vf, oacc[t]);
            }
        }
    }

    // normalize and write bf16 [B, N, D] with head merge (d = h*64 + hd)
    const int b = bh / 12, h = bh - b * 12;
    #pragma unroll
    for (int j = 0; j < 8; ++j) {
        const float inv  = 1.f / row_sum[j];
        const int   nrow = qrow0 + j + 8*hi;
        #pragma unroll
        for (int t = 0; t < 4; ++t) {
            const int d = h*64 + t*16 + l15;
            outA[((size_t)(b*2048 + nrow)) * 768 + d] = f32_to_bf16(oacc[t][j] * inv);
        }
    }
}

// ============================== launch =====================================
extern "C" void kernel_launch(void* const* d_in, const int* in_sizes, int n_in,
                              void* d_out, int out_size, void* d_ws, size_t ws_size,
                              hipStream_t stream) {
    const float* x      = (const float*)d_in[0];   // [2,2048,768]
    const float* Kpsd   = (const float*)d_in[1];   // [2048,2048]
    const float* qkv_w  = (const float*)d_in[2];   // [2304,768]
    const float* qkv_b  = (const float*)d_in[3];   // [2304]
    const float* proj_w = (const float*)d_in[4];   // [768,768]
    const float* proj_b = (const float*)d_in[5];   // [768]
    float* out = (float*)d_out;                    // [2,2048,768] f32

    const size_t HBUF = (size_t)2 * 12 * 2048 * 64;  // 3,145,728 bf16 elems
    u16* qbuf = (u16*)d_ws;
    u16* kbuf = qbuf + HBUF;
    u16* vbuf = kbuf + HBUF;
    u16* abuf = vbuf + HBUF;                          // [2,2048,768] bf16

    // 1) QKV projection + split (M=4096, NC=2304, K=768)
    gemm_bf16<0><<<dim3(2304/128, 4096/128), 256, 0, stream>>>(
        (const void*)x, qkv_w, qkv_b, qbuf, kbuf, vbuf, nullptr);

    // 2) flash attention with multiplicative PSD kernel
    attn_kernel<<<dim3(2048/128, 24), 256, 0, stream>>>(
        qbuf, kbuf, vbuf, Kpsd, abuf);

    // 3) output projection (M=4096, NC=768, K=768), f32 out
    gemm_bf16<1><<<dim3(768/128, 4096/128), 256, 0, stream>>>(
        (const void*)abuf, proj_w, proj_b, nullptr, nullptr, nullptr, out);
}
